// CalculateAttention_7722351198463
// MI455X (gfx1250) — compile-verified, matrix operand order checked by reading, not run on hardware
//
#include <hip/hip_runtime.h>
#include <hip/hip_bf16.h>
#include <math.h>

// CDNA5 / gfx1250 dual-stream flash attention.
// Wave32, v_wmma_f32_16x16x32_bf16. K/V staged to LDS pre-converted to bf16 and
// pre-swizzled into the exact WMMA B-fragment layout (fragment loads are pure
// ds_load_b128). Row max reduced with DPP butterflies; row sums computed by an
// extra WMMA against an all-ones fragment (no lane shuffles on the sum path).

typedef __attribute__((ext_vector_type(16))) __bf16 v16bf;
typedef __attribute__((ext_vector_type(8)))  float  v8f;

#define S_DIM   1024
#define D_DIM   64
#define Q_TILE  128     // query rows per workgroup (8 waves x 16)
#define K_BLK   64      // key rows per inner iteration
#define NWAVES  8
#define LOG2E   1.44269504088896340736f

__device__ __forceinline__ v8f wmma_bf16(v16bf a, v16bf b, v8f c) {
  return __builtin_amdgcn_wmma_f32_16x16x32_bf16(false, a, false, b, (short)0, c,
                                                 false, false);
}

__device__ __forceinline__ unsigned pack_bf16(float a, float b) {
  __bf16 x = (__bf16)a, y = (__bf16)b;
  unsigned short ux = __builtin_bit_cast(unsigned short, x);
  unsigned short uy = __builtin_bit_cast(unsigned short, y);
  return (unsigned)ux | ((unsigned)uy << 16);
}

__device__ __forceinline__ v16bf load_frag(const unsigned* p) {
  union { uint4 q[2]; v16bf bf; } u;
  const uint4* v = (const uint4*)p;
  u.q[0] = v[0];                // ds_load_b128
  u.q[1] = v[1];                // ds_load_b128
  return u.bf;
}

// Butterfly max over the 16-lane row group using DPP (stays on the VALU pipe).
template <int CTRL>
__device__ __forceinline__ float dpp_fmax(float x) {
  int xi = __builtin_bit_cast(int, x);
  int yi = __builtin_amdgcn_update_dpp(xi, xi, CTRL, 0xf, 0xf, true);
  return fmaxf(x, __builtin_bit_cast(float, yi));
}
__device__ __forceinline__ float rowmax16(float x) {
  x = dpp_fmax<0xB1>(x);    // quad_perm [1,0,3,2]  (xor 1)
  x = dpp_fmax<0x4E>(x);    // quad_perm [2,3,0,1]  (xor 2)
  x = dpp_fmax<0x141>(x);   // row_half_mirror      (pairs quads 0-3 / 4-7)
  x = dpp_fmax<0x140>(x);   // row_mirror           (pairs halves 0-7 / 8-15)
  return x;
}

__global__ __launch_bounds__(256, 1)
void fattn2_kernel(const float* __restrict__ Qx, const float* __restrict__ Kx,
                   const float* __restrict__ Vx, const float* __restrict__ Qy,
                   const float* __restrict__ Ky, const float* __restrict__ Vy,
                   float* __restrict__ O1, float* __restrict__ O2) {
  // Packed bf16 fragment buffers.
  // K frag = nh*2 + ch (nh: 16-key tile 0..3, ch: head-dim chunk 0..1).
  // V frag = t*2 + kc  (t: d tile 0..3, kc: 32-key chunk 0..1).
  __shared__ __attribute__((aligned(16))) unsigned sKf[2][8][32][8];   // 16 KB
  __shared__ __attribute__((aligned(16))) unsigned sVf[2][8][32][8];   // 16 KB
  __shared__ __attribute__((aligned(16))) unsigned sPf[NWAVES][2][32][8]; // 16 KB

  const int tid  = threadIdx.x;
  const int lane = tid & 31;
  const int wave = tid >> 5;
  const int m    = lane & 15;
  const int hi   = lane >> 4;

  const int    bh   = blockIdx.y;                 // b*H + h
  const size_t base = (size_t)bh * S_DIM * D_DIM; // [B,H,S,D] row-major
  const float* qx = Qx + base; const float* kx = Kx + base; const float* vx = Vx + base;
  const float* qy = Qy + base; const float* ky = Ky + base; const float* vy = Vy + base;

  const int qrow0 = blockIdx.x * Q_TILE + wave * 16;
  const float qscale = 0.5f * 0.125f * LOG2E;     // log2-domain softmax

  // ---- Q fragments (A-layout), loaded once, pre-scaled ----
  v16bf aqx[2], aqy[2];
#pragma unroll
  for (int c = 0; c < 2; ++c) {
    v16bf tx{}, ty{};
#pragma unroll
    for (int e = 0; e < 16; ++e) {
      const int k = ((e >> 3) << 4) + hi * 8 + (e & 7);
      const size_t off = (size_t)(qrow0 + m) * D_DIM + c * 32 + k;
      tx[e] = (__bf16)(qx[off] * qscale);
      ty[e] = (__bf16)(qy[off] * qscale);
    }
    aqx[c] = tx; aqy[c] = ty;
  }

  // All-ones B fragment: rowsum(P) via WMMA.
  v16bf onesf;
#pragma unroll
  for (int e = 0; e < 16; ++e) onesf[e] = (__bf16)1.0f;

  v8f ox[4], oy[4], lacc = v8f{};
#pragma unroll
  for (int t = 0; t < 4; ++t) { ox[t] = v8f{}; oy[t] = v8f{}; }
  float mrow[8];
#pragma unroll
  for (int r = 0; r < 8; ++r) mrow[r] = -3.0e38f;

  const int nkb = S_DIM / K_BLK;
  for (int kb = 0; kb < nkb; ++kb) {
    __syncthreads();
    // ---- stage K/V block: convert + swizzle into B-fragment layout ----
    for (int i = tid; i < (K_BLK * D_DIM) / 2; i += 256) {
      { // K: pair along head dim (consecutive d, same key)
        const int key = i >> 5, dp = i & 31, d0 = dp << 1;
        const float2 cx = ((const float2*)(kx + (size_t)(kb * K_BLK + key) * D_DIM))[dp];
        const float2 cy = ((const float2*)(ky + (size_t)(kb * K_BLK + key) * D_DIM))[dp];
        const int ch = d0 >> 5, kd = d0 & 31;
        const int ht = (kd >> 3) & 1;
        const int e0 = (kd & 7) | ((kd >> 4) << 3);       // even
        const int lt = (key & 15) | (ht << 4);
        const int fg = ((key >> 4) << 1) | ch;            // nh*2 + ch
        sKf[0][fg][lt][e0 >> 1] = pack_bf16(cx.x, cx.y);
        sKf[1][fg][lt][e0 >> 1] = pack_bf16(cy.x, cy.y);
      }
      { // V: pair along key dim (consecutive keys, same d)
        const int d = i & 63, kp = i >> 6, k0 = kp << 1;
        const size_t g0 = (size_t)(kb * K_BLK + k0) * D_DIM + d;
        const float ax = vx[g0], bx = vx[g0 + D_DIM];
        const float ay = vy[g0], by = vy[g0 + D_DIM];
        const int kc = k0 >> 5, k32 = k0 & 31;
        const int ht = (k32 >> 3) & 1;
        const int e0 = (k32 & 7) | ((k32 >> 4) << 3);     // even
        const int lt = (d & 15) | (ht << 4);
        const int fg = ((d >> 4) << 1) | kc;              // t*2 + kc
        sVf[0][fg][lt][e0 >> 1] = pack_bf16(ax, bx);
        sVf[1][fg][lt][e0 >> 1] = pack_bf16(ay, by);
      }
      if (kb + 1 < nkb) { // pull next block toward the caches
        const size_t gn = (size_t)((kb + 1) * K_BLK + (i >> 5)) * D_DIM + ((i & 31) << 1);
        __builtin_prefetch(kx + gn, 0, 1); __builtin_prefetch(ky + gn, 0, 1);
        __builtin_prefetch(vx + gn, 0, 1); __builtin_prefetch(vy + gn, 0, 1);
      }
    }
    __syncthreads();

    // ---- scores: (Qx*Kx^T + Qy*Ky^T) over 4 key tiles of 16 ----
    v8f sacc[4];
#pragma unroll
    for (int nh = 0; nh < 4; ++nh) {
      v8f c{};
#pragma unroll
      for (int ch = 0; ch < 2; ++ch) {
        c = wmma_bf16(aqx[ch], load_frag(&sKf[0][nh * 2 + ch][lane][0]), c);
        c = wmma_bf16(aqy[ch], load_frag(&sKf[1][nh * 2 + ch][lane][0]), c);
      }
      sacc[nh] = c;
    }

    // ---- online softmax (log2 domain), max via DPP butterfly ----
#pragma unroll
    for (int r = 0; r < 8; ++r) {
      float mx = fmaxf(fmaxf(sacc[0][r], sacc[1][r]),
                       fmaxf(sacc[2][r], sacc[3][r]));
      mx = rowmax16(mx);
      const float mnew  = fmaxf(mrow[r], mx);
      const float alpha = exp2f(mrow[r] - mnew);
      mrow[r] = mnew;
#pragma unroll
      for (int nh = 0; nh < 4; ++nh) sacc[nh][r] = exp2f(sacc[nh][r] - mnew);
      lacc[r] *= alpha;
#pragma unroll
      for (int t = 0; t < 4; ++t) { ox[t][r] *= alpha; oy[t][r] *= alpha; }
    }

    // ---- P: C-layout -> packed bf16 A-layout via per-wave LDS bounce ----
    __bf16* pw0 = (__bf16*)&sPf[wave][0][0][0];
    __bf16* pw1 = (__bf16*)&sPf[wave][1][0][0];
#pragma unroll
    for (int r = 0; r < 8; ++r) {
      const int row = r + hi * 8;
      const int ht  = (m >> 3) & 1;          // same for cols m and m+16
      const int lt  = row | (ht << 4);
      const int eb  = m & 7;
      pw0[lt * 16 + eb]     = (__bf16)sacc[0][r];   // keys  0..15
      pw0[lt * 16 + eb + 8] = (__bf16)sacc[1][r];   // keys 16..31
      pw1[lt * 16 + eb]     = (__bf16)sacc[2][r];   // keys 32..47
      pw1[lt * 16 + eb + 8] = (__bf16)sacc[3][r];   // keys 48..63
    }
    __syncthreads();
    v16bf pa[2];
    pa[0] = load_frag(&sPf[wave][0][lane][0]);
    pa[1] = load_frag(&sPf[wave][1][lane][0]);

    // ---- row sums via WMMA against ones (replaces shuffle reduction) ----
    lacc = wmma_bf16(pa[0], onesf, lacc);
    lacc = wmma_bf16(pa[1], onesf, lacc);

    // ---- O += P * V for both streams ----
#pragma unroll
    for (int t = 0; t < 4; ++t) {
#pragma unroll
      for (int kc = 0; kc < 2; ++kc) {
        ox[t] = wmma_bf16(pa[kc], load_frag(&sVf[0][t * 2 + kc][lane][0]), ox[t]);
        oy[t] = wmma_bf16(pa[kc], load_frag(&sVf[1][t * 2 + kc][lane][0]), oy[t]);
      }
    }
  }

  // ---- epilogue: normalize + store ----
#pragma unroll
  for (int r = 0; r < 8; ++r) {
    const float inv = 1.0f / lacc[r];
    const size_t row = (size_t)(qrow0 + r + hi * 8) * D_DIM;
#pragma unroll
    for (int t = 0; t < 4; ++t) {
      O1[base + row + t * 16 + m] = ox[t][r] * inv;
      O2[base + row + t * 16 + m] = oy[t][r] * inv;
    }
  }
}

extern "C" void kernel_launch(void* const* d_in, const int* in_sizes, int n_in,
                              void* d_out, int out_size, void* d_ws, size_t ws_size,
                              hipStream_t stream) {
  const float* Qx = (const float*)d_in[0];
  const float* Kx = (const float*)d_in[1];
  const float* Vx = (const float*)d_in[2];
  const float* Qy = (const float*)d_in[3];
  const float* Ky = (const float*)d_in[4];
  const float* Vy = (const float*)d_in[5];

  const int B = 4, H = 16;
  float* O1 = (float*)d_out;
  float* O2 = O1 + (size_t)B * H * S_DIM * D_DIM;

  dim3 grid(S_DIM / Q_TILE, B * H);
  fattn2_kernel<<<grid, 256, 0, stream>>>(Qx, Kx, Vx, Qy, Ky, Vy, O1, O2);
}
